// FeedForwardQuantum_65481071398154
// MI455X (gfx1250) — compile-verified, matrix operand order checked by reading, not run on hardware
//
#include <hip/hip_runtime.h>

// Problem sizes (from reference): B=8, S=2048, E=768, FF=3072, Q=8
#define EQ   768
#define FFQ  3072
#define QQ   8
#define MTOT 16384            // B*S tokens

#define TM   128              // workgroup M tile (tokens)
#define TN   256              // workgroup N tile (embed cols)
#define FFC  128              // FF chunk staged in LDS
#define LDH  (FFC + 8)        // hLDS row stride (elements), +16B pad vs bank conflicts
#define LDM  40               // measLDS row stride (elements)

typedef __attribute__((ext_vector_type(16))) __bf16        v16bf;
typedef __attribute__((ext_vector_type(8)))  float         v8f;
typedef __attribute__((ext_vector_type(4)))  unsigned int  v4u;

union Frag {
  v16bf        bf;
  v4u          q[2];
  unsigned int u[8];
};

// ---------------------------------------------------------------- prologue
__global__ __launch_bounds__(256) void ffq_cvt_bf16(const float* __restrict__ s,
                                                    __bf16* __restrict__ d, int n) {
  int i = blockIdx.x * blockDim.x + threadIdx.x;
  if (i < n) d[i] = (__bf16)s[i];
}

// ---------------------------------------------------------------- main fused kernel
__global__ __launch_bounds__(256) void ffq_main(const float* __restrict__ x,
                                                const float* __restrict__ phi,
                                                const float* __restrict__ b1,
                                                const float* __restrict__ b2,
                                                const __bf16* __restrict__ w1b,
                                                const __bf16* __restrict__ w2b,
                                                float* __restrict__ out) {
  __shared__ __bf16 measL[TM * LDM];   // 10 KB : meas padded to K=32 (zeros for k>=8)
  __shared__ __bf16 hL[TM * LDH];      // 34 KB : relu(meas@w1^T+b1) chunk, bf16

  const int tid   = threadIdx.x;
  const int lane  = tid & 31;          // wave32
  const int wave  = tid >> 5;          // 0..7
  const int lhalf = lane >> 4;         // 0/1 : K-half select per ISA frag layout
  const int l16   = lane & 15;
  const int wm    = wave >> 2;         // 0..1 : wave row (64 tokens each)
  const int wn    = wave & 3;          // 0..3 : wave col (64 embeds each)

  const int Mbase = blockIdx.y * TM;
  const int Nbase = blockIdx.x * TN;

  // ---- phase 0: meas[m][k] = k<8 ? cos(x[tok,k]+phi[k]) : 0  (bf16, K padded to 32)
  for (int idx = tid; idx < TM * 32; idx += 256) {
    int m = idx >> 5, k = idx & 31;
    float v = 0.0f;
    if (k < QQ) v = __builtin_cosf(x[(size_t)(Mbase + m) * EQ + k] + phi[k]);
    measL[m * LDM + k] = (__bf16)v;
  }
  __syncthreads();

  // A-fragment of meas for this wave's phase-1 m-tile (rows wave*16 .. +15), reused all chunks.
  // 16-bit A 16x32 layout: lanes 0-15 hold K[0..7]|K[16..23], lanes 16-31 hold K[8..15]|K[24..31].
  Frag ameas;
  {
    const __bf16* p = &measL[((wave << 4) + l16) * LDM + (lhalf << 3)];
    ameas.q[0] = *(const v4u*)p;
    ameas.q[1] = *(const v4u*)(p + 16);
  }

  v8f acc[4][4] = {};   // 64x64 f32 accumulator block per wave

  for (int c = 0; c < FFQ / FFC; ++c) {
    if (c) __syncthreads();   // phase-2 readers of previous chunk must finish

    // ---- phase 1: hL[TM][FFC] = relu(meas @ w1^T + b1) via WMMA, wave w owns m-tile w
    for (int nt = 0; nt < FFC / 16; ++nt) {
      Frag bw;
      #pragma unroll
      for (int r = 0; r < 8; ++r) bw.u[r] = 0u;
      const int f = c * FFC + nt * 16 + l16;      // global FF column
      // B 32x16 layout mirror: lanes 0-15 / VGPR0-3 carry K=0..7 (only nonzero K of meas)
      if (lhalf == 0) bw.q[0] = *(const v4u*)(w1b + (size_t)f * QQ);
      v8f d = {};
      d = __builtin_amdgcn_wmma_f32_16x16x32_bf16(false, ameas.bf, false, bw.bf,
                                                  (short)0, d, false, false);
      const float bias = b1[f];
      #pragma unroll
      for (int r = 0; r < 8; ++r) {
        float hv = d[r] + bias;
        hv = hv > 0.0f ? hv : 0.0f;
        const int m = (wave << 4) + r + (lhalf << 3);   // C/D layout: M = r + 8*lhalf
        hL[m * LDH + nt * 16 + l16] = (__bf16)hv;
      }
    }
    __syncthreads();

    // ---- phase 2: acc += h_chunk @ w2_chunk^T   (K = FFC in steps of 32)
    #pragma unroll
    for (int s = 0; s < FFC / 32; ++s) {
      const int k0 = s * 32;
      Frag A[4], Bf[4];
      #pragma unroll
      for (int i = 0; i < 4; ++i) {
        const __bf16* p = &hL[((wm * 4 + i) * 16 + l16) * LDH + k0 + (lhalf << 3)];
        A[i].q[0] = *(const v4u*)p;          // ds_load_b128
        A[i].q[1] = *(const v4u*)(p + 16);
      }
      #pragma unroll
      for (int j = 0; j < 4; ++j) {
        const int n = Nbase + wn * 64 + j * 16 + l16;
        const __bf16* p = w2b + (size_t)n * FFQ + c * FFC + k0 + (lhalf << 3);
        Bf[j].q[0] = *(const v4u*)p;         // global_load_b128 (L2-resident w2)
        Bf[j].q[1] = *(const v4u*)(p + 16);
      }
      #pragma unroll
      for (int i = 0; i < 4; ++i)
        #pragma unroll
        for (int j = 0; j < 4; ++j)
          acc[i][j] = __builtin_amdgcn_wmma_f32_16x16x32_bf16(false, A[i].bf, false, Bf[j].bf,
                                                              (short)0, acc[i][j], false, false);
    }
  }

  // ---- epilogue: out = acc + b2
  #pragma unroll
  for (int j = 0; j < 4; ++j) {
    const int n = Nbase + wn * 64 + j * 16 + l16;
    const float bias = b2[n];
    #pragma unroll
    for (int i = 0; i < 4; ++i) {
      const int mB = Mbase + wm * 64 + i * 16 + (lhalf << 3);
      #pragma unroll
      for (int r = 0; r < 8; ++r)
        out[(size_t)(mB + r) * EQ + n] = acc[i][j][r] + bias;
    }
  }
}

// ---------------------------------------------------------------- launcher
extern "C" void kernel_launch(void* const* d_in, const int* in_sizes, int n_in,
                              void* d_out, int out_size, void* d_ws, size_t ws_size,
                              hipStream_t stream) {
  (void)in_sizes; (void)n_in; (void)out_size; (void)ws_size;
  const float* x   = (const float*)d_in[0];
  const float* phi = (const float*)d_in[1];
  const float* w1  = (const float*)d_in[2];
  const float* b1  = (const float*)d_in[3];
  const float* w2  = (const float*)d_in[4];
  const float* b2  = (const float*)d_in[5];
  float* out = (float*)d_out;

  // workspace: bf16 copies of w2 [E,FF] and w1 [FF,Q]
  __bf16* w2b = (__bf16*)d_ws;
  __bf16* w1b = (__bf16*)((char*)d_ws + (size_t)EQ * FFQ * sizeof(__bf16));

  const int n2 = EQ * FFQ;
  ffq_cvt_bf16<<<(n2 + 255) / 256, 256, 0, stream>>>(w2, w2b, n2);
  const int n1 = FFQ * QQ;
  ffq_cvt_bf16<<<(n1 + 255) / 256, 256, 0, stream>>>(w1, w1b, n1);

  dim3 grid(EQ / TN, MTOT / TM);   // (3, 128)
  ffq_main<<<grid, 256, 0, stream>>>(x, phi, b1, b2, w1b, w2b, out);
}
